// EncoderEstimator_47854525612384
// MI455X (gfx1250) — matrix operand order repaired and checked by reading, not compiled, not run on hardware
//
#include <hip/hip_runtime.h>
#include <hip/hip_bf16.h>

typedef __attribute__((ext_vector_type(16))) _Float16 v16h;
typedef __attribute__((ext_vector_type(8)))  float    v8f;

#define IMG  256
#define KM   64
#define CCH  32
#define WIN  5
#define HALFW 2
#define ROWSTRIDE 36   // 32 channels + 1 sum-of-squares + 3 pad floats (144 B, 16B-aligned)

// ---------------------------------------------------------------------------
// Kernel 1: conv encoder as im2col GEMM via WMMA f32 <- f16*f16
//   M = 4096 output pixels, N = 32 channels, K = 48 (padded to 64)
//   One wave per 16x16 output tile: 256 M-tiles x 2 N-tiles = 512 waves.
//   Output written pixel-major with padded rows: keymap_t[pix*36 + ch]
// ---------------------------------------------------------------------------
__global__ __launch_bounds__(256)
void conv_wmma_kernel(const float* __restrict__ query,
                      const float* __restrict__ convw,
                      const float* __restrict__ convb,
                      float* __restrict__ keymap_t)
{
    const int lane   = threadIdx.x & 31;
    const int waveId = (blockIdx.x * (blockDim.x >> 5)) + (threadIdx.x >> 5);
    const int mtile  = waveId >> 1;        // 0..255
    const int ntile  = waveId & 1;         // 0..1

    // A-matrix row handled by this lane (16x32 f16 A-frag layout, ISA 7.12.2)
    const int mrow = mtile * 16 + (lane & 15);
    const int oy   = mrow >> 6;            // pixel y (0..63)
    const int ox   = mrow & 63;            // pixel x (0..63)
    const int hiK  = (lane >= 16) ? 8 : 0; // A: upper half-wave holds K+8
    const int hiKB = (lane >= 16) ? 16 : 0;// B: same K order as A (upper half-wave holds K+8)
    const int ncol = ntile * 16 + (lane & 15);

    v8f acc = {};

    #pragma unroll
    for (int kc = 0; kc < 2; ++kc) {       // two K-chunks of 32
        // ---- A fragment: element i -> VGPR i/2, half i%2 ----
        v16h a;
        #pragma unroll
        for (int i = 0; i < 16; ++i) {
            const int v = i >> 1, h = i & 1;
            const int kk = (v < 4) ? (hiK + 2 * v + h)
                                   : (16 + hiK + 2 * (v - 4) + h);
            const int k  = kc * 32 + kk;   // global K index 0..63
            float val = 0.0f;
            if (k < 48) {
                const int c  = k >> 4;     // input channel 0..2
                const int r  = k & 15;
                const int kh = r >> 2;
                const int kw = r & 3;
                val = query[c * (IMG * IMG) + (oy * 4 + kh) * IMG + (ox * 4 + kw)];
            }
            a[i] = (_Float16)val;
        }
        // ---- B fragment: 32x16, element i -> VGPR i/2, half i%2 ----
        v16h b;
        #pragma unroll
        for (int i = 0; i < 16; ++i) {
            const int v = i >> 1, h = i & 1;
            const int k = kc * 32 + ((v < 4) ? (hiK + 2 * v + h) : (16 + hiK + 2 * (v - 4) + h));  // 0..63
            float val = 0.0f;
            if (k < 48) val = convw[ncol * 48 + k];    // OIHW flat: w[n][k]
            b[i] = (_Float16)val;
        }
        acc = __builtin_amdgcn_wmma_f32_16x16x32_f16(
                  false, a, false, b, (short)0, acc, false, false);
    }

    // ---- D store: VGPR r -> M = r (+8 for upper half-wave), N = lane&15 ----
    const float bias = convb[ncol];
    const int mofs = (lane >= 16) ? 8 : 0;
    #pragma unroll
    for (int r = 0; r < 8; ++r) {
        const int pix = mtile * 16 + r + mofs;
        keymap_t[pix * ROWSTRIDE + ncol] = acc[r] + bias;
    }
}

// ---------------------------------------------------------------------------
// Kernel 1b: per-pixel sum-of-squares, stored at row offset 32.
//   4096 pixels, one thread each. Enables dist^2 = Sp2 - 2*dot + Sk2 in the
//   hot loop (halves its VALU work).
// ---------------------------------------------------------------------------
__global__ __launch_bounds__(256)
void sumsq_kernel(float* __restrict__ keymap_t)
{
    const int pix = blockIdx.x * blockDim.x + threadIdx.x;
    if (pix >= KM * KM) return;
    const float4* row = reinterpret_cast<const float4*>(keymap_t + (size_t)pix * ROWSTRIDE);
    float s = 0.0f;
    #pragma unroll
    for (int i = 0; i < 8; ++i) {
        const float4 p = row[i];
        s += p.x * p.x + p.y * p.y + p.z * p.z + p.w * p.w;
    }
    keymap_t[(size_t)pix * ROWSTRIDE + CCH] = s;
}

// ---------------------------------------------------------------------------
// Kernel 2: per-keypoint 5x5 weighted L2 distance + argmin (bandwidth-bound).
//   keymap_t is 576 KB -> resident in 192 MB L2; gathers hit L2.
//   One thread per keypoint; memory key held in registers (8 x float4).
//   Inner loop is a pure dot product: ||p-k||^2 = Sp2 - 2*(p.k) + Sk2.
// ---------------------------------------------------------------------------
__global__ __launch_bounds__(256)
void dist_argmin_kernel(const float* __restrict__ memory_keys,
                        const int*   __restrict__ keypoints,
                        const float* __restrict__ keymap_t,
                        int*   __restrict__ out_kp,
                        float* __restrict__ out_wdist,
                        int n_kp)
{
    const int n = blockIdx.x * blockDim.x + threadIdx.x;
    if (n >= n_kp) return;

    const int kx = keypoints[2 * n + 0];
    const int ky = keypoints[2 * n + 1];

    // load 32-float memory key into registers and its sum of squares
    float4 mk[8];
    const float4* mkp = reinterpret_cast<const float4*>(memory_keys + (size_t)n * CCH);
    float ksum = 0.0f;
    #pragma unroll
    for (int i = 0; i < 8; ++i) {
        mk[i] = mkp[i];
        ksum += mk[i].x * mk[i].x + mk[i].y * mk[i].y +
                mk[i].z * mk[i].z + mk[i].w * mk[i].w;
    }

    // prefetch first patch row (gfx1250 global_prefetch_b8 path)
    __builtin_prefetch(keymap_t + (size_t)((ky - HALFW) * KM + (kx - HALFW)) * ROWSTRIDE, 0, 1);

    float best    = 3.4e38f;
    int   bestIdx = 0;

    #pragma unroll
    for (int dy = 0; dy < WIN; ++dy) {
        const int y = ky + dy - HALFW;      // in [0,63] by construction
        const float cy = fabsf((float)(dy - HALFW));
        #pragma unroll
        for (int dx = 0; dx < WIN; ++dx) {
            const int x = kx + dx - HALFW;
            const float* rowp = keymap_t + (size_t)(y * KM + x) * ROWSTRIDE;
            const float4* pp = reinterpret_cast<const float4*>(rowp);
            float dot = 0.0f;
            #pragma unroll
            for (int i = 0; i < 8; ++i) {
                const float4 p = pp[i];
                dot += p.x * mk[i].x + p.y * mk[i].y +
                       p.z * mk[i].z + p.w * mk[i].w;
            }
            const float psum = rowp[CCH];
            const float d2   = fmaxf(psum + ksum - 2.0f * dot, 0.0f);
            const float cx = fabsf((float)(dx - HALFW));
            const float w  = 0.5f + 0.1f * fmaxf(cx, cy);
            const float wd = sqrtf(d2) * w;
            const int idx  = dy * WIN + dx;
            out_wdist[(size_t)n * (WIN * WIN) + idx] = wd;
            if (wd < best) { best = wd; bestIdx = idx; }  // first-min tie rule
        }
    }

    const int min_x = bestIdx % WIN;
    const int min_y = bestIdx / WIN;
    out_kp[2 * n + 0] = kx + (HALFW - min_x);
    out_kp[2 * n + 1] = ky + (HALFW - min_y);
}

// ---------------------------------------------------------------------------
extern "C" void kernel_launch(void* const* d_in, const int* in_sizes, int n_in,
                              void* d_out, int out_size, void* d_ws, size_t ws_size,
                              hipStream_t stream)
{
    const float* query  = (const float*)d_in[0];   // (1,3,256,256)
    const float* convw  = (const float*)d_in[1];   // (32,3,4,4)
    const float* convb  = (const float*)d_in[2];   // (32,)
    const float* mkeys  = (const float*)d_in[3];   // (N,32)
    const int*   kpts   = (const int*)  d_in[4];   // (N,2)

    const int n_kp = in_sizes[4] / 2;

    float* keymap_t = (float*)d_ws;                // 4096 * 36 floats = 576 KB

    // outputs: new_keypoints (N,2) int32, then wdist (N,5,5) float32
    int*   out_kp = (int*)d_out;
    float* out_wd = (float*)d_out + (size_t)2 * n_kp;

    // conv as WMMA GEMM: 512 tiles, 8 waves (256 thr) per block -> 64 blocks
    conv_wmma_kernel<<<64, 256, 0, stream>>>(query, convw, convb, keymap_t);

    // per-pixel sum of squares (4096 pixels)
    sumsq_kernel<<<(KM * KM + 255) / 256, 256, 0, stream>>>(keymap_t);

    // distance/argmin pass: one thread per keypoint
    const int blk = 256;
    dist_argmin_kernel<<<(n_kp + blk - 1) / blk, blk, 0, stream>>>(
        mkeys, kpts, keymap_t, out_kp, out_wd, n_kp);
}
